// DistributedChunkedHybridContrastive_62388694942233
// MI455X (gfx1250) — matrix operand order repaired and checked by reading, not compiled, not run on hardware
//
#include <hip/hip_runtime.h>
#include <hip/hip_bf16.h>

typedef __attribute__((ext_vector_type(16))) _Float16 v16h;
typedef __attribute__((ext_vector_type(8)))  _Float16 v8h;
typedef __attribute__((ext_vector_type(8)))  float    v8f;
typedef __attribute__((ext_vector_type(4)))  int      v4i;

#define CN 8192            // rows/cols
#define CD 128             // feature dim
#define TILE_J 16          // column tile (WMMA N)
#define NTILES (CN / TILE_J)       // 512
#define CHUNKS (CD / 32)           // 4 K-chunks of 32
#define WAVES 4
#define THREADS (WAVES * 32)       // 128
#define ROWS_PER_WG (WAVES * 16)   // 64
#define WGS_PER_DIR (CN / ROWS_PER_WG) // 128
#define LDS_PITCH 136      // 128 halves + 8 pad (272B row stride -> conflict-free)

#define CAT16(a, b) __builtin_shufflevector(a, b, 0,1,2,3,4,5,6,7,8,9,10,11,12,13,14,15)

// ---- CDNA5 async global->LDS path (guarded; falls back to sync staging) ----
// Builtin signature (per hipcc diagnostic): (int4 AS1*, int4 AS3*, imm offset, imm cpol)
#if __has_builtin(__builtin_amdgcn_global_load_async_to_lds_b128)
#define HAVE_ASYNC_LDS 1
#define AS1P(p) ((__attribute__((address_space(1))) v4i*)(void*)(p))
#define AS3P(p) ((__attribute__((address_space(3))) v4i*)(void*)(p))
#else
#define HAVE_ASYNC_LDS 0
#endif

#if __has_builtin(__builtin_amdgcn_s_wait_asynccnt)
#define WAIT_ASYNC(n) __builtin_amdgcn_s_wait_asynccnt(n)
#elif HAVE_ASYNC_LDS
#define WAIT_ASYNC(n) asm volatile("s_wait_asynccnt %0" :: "i"(n) : "memory")
#else
#define WAIT_ASYNC(n)
#endif

// ---------------------------------------------------------------------------
// Pass 1: split f32 features into hi/lo f16 pairs (x = hi + lo, |lo| ~ 2^-11 |x|)
// so the GEMM can run on v_wmma_f32_16x16x32_f16 with ~f32 accuracy.
// ---------------------------------------------------------------------------
__global__ void convert_hilo(const float* __restrict__ fa,
                             const float* __restrict__ fb,
                             _Float16* __restrict__ ah, _Float16* __restrict__ al,
                             _Float16* __restrict__ bh, _Float16* __restrict__ bl) {
    size_t i = (size_t)blockIdx.x * blockDim.x + threadIdx.x;
    if (i < (size_t)CN * CD) {
        float xa = fa[i];
        _Float16 ha = (_Float16)xa;
        ah[i] = ha;
        al[i] = (_Float16)(xa - (float)ha);
        float xb = fb[i];
        _Float16 hb = (_Float16)xb;
        bh[i] = hb;
        bl[i] = (_Float16)(xb - (float)hb);
    }
}

// split-f16 GEMM tile + fold into running softmax/positive stats
__device__ __forceinline__ void compute_tile(
        const _Float16 (&lH)[TILE_J][LDS_PITCH],
        const _Float16 (&lL)[TILE_J][LDS_PITCH],
        const v16h (&a_hi)[CHUNKS], const v16h (&a_lo)[CHUNKS],
        const int* __restrict__ targets, const int (&trow)[8],
        float Tval, int j0, int l16, int h,
        float (&sacc)[8], float (&wacc)[8], float (&cacc)[8]) {
    v8f acc0 = {}; v8f acc1 = {};
    #pragma unroll
    for (int c = 0; c < CHUNKS; ++c) {
        // B layout: lane l16 = column N (= feature row j0+l16); K = c*32 + 8*h + 0..7, then +16
        const int koff = c * 32 + h * 8;
        v8h b0 = *(const v8h*)(&lH[l16][koff]);
        v8h b1 = *(const v8h*)(&lH[l16][koff + 16]);
        v16h bhv = CAT16(b0, b1);
        v8h d0 = *(const v8h*)(&lL[l16][koff]);
        v8h d1 = *(const v8h*)(&lL[l16][koff + 16]);
        v16h blv = CAT16(d0, d1);

        v8f& acc = (c & 1) ? acc1 : acc0;
        acc = __builtin_amdgcn_wmma_f32_16x16x32_f16(false, a_hi[c], false, bhv, (short)0, acc, false, false);
        acc = __builtin_amdgcn_wmma_f32_16x16x32_f16(false, a_hi[c], false, blv, (short)0, acc, false, false);
        acc = __builtin_amdgcn_wmma_f32_16x16x32_f16(false, a_lo[c], false, bhv, (short)0, acc, false, false);
    }
    // Unit-norm features => logits <= T, so fixed shift M = T is a valid
    // logsumexp shift (no online-max rescaling needed).
    const int tj = targets[j0 + l16];
    #pragma unroll
    for (int r = 0; r < 8; ++r) {
        float x = Tval * (acc0[r] + acc1[r]);
        sacc[r] += __expf(x - Tval);
        if (tj == trow[r]) { wacc[r] += x; cacc[r] += 1.0f; }
    }
}

// ---------------------------------------------------------------------------
// Pass 2: fused GEMM + masked-positive sum + fixed-shift logsumexp.
// Blocks [0,128) do fa@fb^T rows, blocks [128,256) do fb@fa^T rows.
// ---------------------------------------------------------------------------
__global__ __launch_bounds__(THREADS)
void contrastive_fused(const _Float16* __restrict__ fah,
                       const _Float16* __restrict__ fal,
                       const _Float16* __restrict__ fbh,
                       const _Float16* __restrict__ fbl,
                       const int* __restrict__ targets,
                       const float* __restrict__ Tp,
                       float* __restrict__ partials) {
    __shared__ _Float16 ldsH[2][TILE_J][LDS_PITCH];
    __shared__ _Float16 ldsL[2][TILE_J][LDS_PITCH];
    __shared__ float wred[WAVES];

    const int dir = blockIdx.x / WGS_PER_DIR;
    const int wg  = blockIdx.x % WGS_PER_DIR;
    const _Float16* Ah = dir ? fbh : fah;
    const _Float16* Al = dir ? fbl : fal;
    const _Float16* Bh = dir ? fah : fbh;
    const _Float16* Bl = dir ? fal : fbl;

    const int tid  = threadIdx.x;
    const int lane = tid & 31;
    const int wave = tid >> 5;
    const int h    = lane >> 4;   // lane half (K-split for A/B, row-split for C)
    const int l16  = lane & 15;

    const float Tval = Tp[0];
    const int rowbase = wg * ROWS_PER_WG + wave * 16;

    // ---- A strip (16 rows x 128) hi/lo into registers, WMMA A layout:
    // lane l16 = row M; halves[0..7] = K = c*32 + 8*h + t; halves[8..15] = +16.
    v16h a_hi[CHUNKS], a_lo[CHUNKS];
    {
        const size_t rb = (size_t)(rowbase + l16) * CD;
        #pragma unroll
        for (int c = 0; c < CHUNKS; ++c) {
            const int k0 = c * 32 + h * 8;
            v8h p0 = *(const v8h*)(Ah + rb + k0);
            v8h p1 = *(const v8h*)(Ah + rb + k0 + 16);
            a_hi[c] = CAT16(p0, p1);
            v8h q0 = *(const v8h*)(Al + rb + k0);
            v8h q1 = *(const v8h*)(Al + rb + k0 + 16);
            a_lo[c] = CAT16(q0, q1);
        }
    }

    // targets of the 8 output rows this lane's accumulator VGPRs map to
    // (C/D layout: VGPR r holds row rowbase + 8*h + r, lane l16 = column)
    int trow[8];
    #pragma unroll
    for (int r = 0; r < 8; ++r) trow[r] = targets[rowbase + 8 * h + r];

    float sacc[8], wacc[8], cacc[8];
    #pragma unroll
    for (int r = 0; r < 8; ++r) { sacc[r] = 0.f; wacc[r] = 0.f; cacc[r] = 0.f; }

#if HAVE_ASYNC_LDS
    // ===================== async double-buffered pipeline =====================
    // Each thread DMAs 4 x 16B of the 16x128 hi/lo tile straight into LDS.
    const int sr = tid >> 3;        // 0..15: staged tile row
    const int sc = (tid & 7) * 8;   // halves; 16B granule, covers cols 0..63 / 64..127

    #define STAGE_ASYNC(tileidx, buf)                                                    \
        do {                                                                             \
            const size_t g = (size_t)((tileidx) * TILE_J + sr) * CD + sc;                \
            __builtin_amdgcn_global_load_async_to_lds_b128(                              \
                AS1P(Bh + g),      AS3P(&ldsH[buf][sr][sc]),      0, 0);                 \
            __builtin_amdgcn_global_load_async_to_lds_b128(                              \
                AS1P(Bh + g + 64), AS3P(&ldsH[buf][sr][sc + 64]), 0, 0);                 \
            __builtin_amdgcn_global_load_async_to_lds_b128(                              \
                AS1P(Bl + g),      AS3P(&ldsL[buf][sr][sc]),      0, 0);                 \
            __builtin_amdgcn_global_load_async_to_lds_b128(                              \
                AS1P(Bl + g + 64), AS3P(&ldsL[buf][sr][sc + 64]), 0, 0);                 \
        } while (0)

    STAGE_ASYNC(0, 0);
    for (int tile = 0; tile < NTILES; ++tile) {
        const int cur = tile & 1;
        // buf[cur^1] was last read at tile-1; the end-of-iteration barrier
        // guarantees every wave is done with it before these writes land.
        if (tile + 1 < NTILES) {
            STAGE_ASYNC(tile + 1, cur ^ 1);
            WAIT_ASYNC(4);   // async loads complete in order: tile's 4 ops done
        } else {
            WAIT_ASYNC(0);
        }
        __syncthreads();     // tile data visible to all waves

        compute_tile(ldsH[cur], ldsL[cur], a_hi, a_lo, targets, trow,
                     Tval, tile * TILE_J, l16, h, sacc, wacc, cacc);

        __syncthreads();     // all waves done reading buf[cur] before overwrite
    }
    #undef STAGE_ASYNC
#else
    // ===================== synchronous fallback =====================
    const int ldr = tid >> 3;        // 0..15: B-tile row staged by this thread
    const int ldc = (tid & 7) * 16;  // 32B segment within the 256B row

    for (int tile = 0; tile < NTILES; ++tile) {
        const int j0 = tile * TILE_J;
        {
            const size_t gb = (size_t)(j0 + ldr) * CD + ldc;
            const uint4* sh = (const uint4*)(Bh + gb);
            uint4 h0 = sh[0], h1 = sh[1];
            const uint4* sl = (const uint4*)(Bl + gb);
            uint4 l0 = sl[0], l1 = sl[1];
            uint4* dh = (uint4*)(&ldsH[0][ldr][ldc]);
            dh[0] = h0; dh[1] = h1;
            uint4* dl = (uint4*)(&ldsL[0][ldr][ldc]);
            dl[0] = l0; dl[1] = l1;
        }
        __syncthreads();

        compute_tile(ldsH[0], ldsL[0], a_hi, a_lo, targets, trow,
                     Tval, j0, l16, h, sacc, wacc, cacc);

        __syncthreads();
    }
#endif

    // reduce each row's stats across the 16 lanes of the half, then finish rows:
    // per_row = -wsum/cnt + (T + log(sumexp))
    float rowsum = 0.f;
    #pragma unroll
    for (int r = 0; r < 8; ++r) {
        float s = sacc[r], w = wacc[r], cnum = cacc[r];
        #pragma unroll
        for (int m = 1; m < 16; m <<= 1) {
            s    += __shfl_xor(s, m, 32);
            w    += __shfl_xor(w, m, 32);
            cnum += __shfl_xor(cnum, m, 32);
        }
        rowsum += -w / cnum + Tval + __logf(s);
    }
    // combine the two lane-halves (rows 0-7 and 8-15 of the strip)
    rowsum += __shfl_xor(rowsum, 16, 32);

    if (lane == 0) wred[wave] = rowsum;
    __syncthreads();
    if (tid == 0) {
        partials[blockIdx.x] = wred[0] + wred[1] + wred[2] + wred[3];
    }
}

// ---------------------------------------------------------------------------
// Pass 3: deterministic fixed-order reduction of the 256 per-WG partials.
// loss = (mean_i + mean_t)/2 = total / (2N)
// ---------------------------------------------------------------------------
__global__ void final_reduce(const float* __restrict__ partials, float* __restrict__ out) {
    __shared__ float buf[2 * WGS_PER_DIR];
    const int t = threadIdx.x;
    buf[t] = partials[t];
    __syncthreads();
    for (int s = WGS_PER_DIR; s > 0; s >>= 1) {
        if (t < s) buf[t] += buf[t + s];
        __syncthreads();
    }
    if (t == 0) out[0] = buf[0] * (1.0f / (2.0f * (float)CN));
}

extern "C" void kernel_launch(void* const* d_in, const int* in_sizes, int n_in,
                              void* d_out, int out_size, void* d_ws, size_t ws_size,
                              hipStream_t stream) {
    (void)in_sizes; (void)n_in; (void)out_size; (void)ws_size;
    const float* Tp      = (const float*)d_in[0];
    const float* fa      = (const float*)d_in[1];
    const float* fb      = (const float*)d_in[2];
    const int*   targets = (const int*)d_in[3];
    // d_in[4] = training flag: unused (forward value only)

    char* ws = (char*)d_ws;
    const size_t fbytes = (size_t)CN * CD * sizeof(_Float16);   // 2 MB each
    _Float16* ah = (_Float16*)(ws + 0 * fbytes);
    _Float16* al = (_Float16*)(ws + 1 * fbytes);
    _Float16* bh = (_Float16*)(ws + 2 * fbytes);
    _Float16* bl = (_Float16*)(ws + 3 * fbytes);
    float* partials = (float*)(ws + 4 * fbytes);

    const int nconv = CN * CD;
    convert_hilo<<<(nconv + 255) / 256, 256, 0, stream>>>(fa, fb, ah, al, bh, bl);
    contrastive_fused<<<2 * WGS_PER_DIR, THREADS, 0, stream>>>(ah, al, bh, bl, targets, Tp, partials);
    final_reduce<<<1, 2 * WGS_PER_DIR, 0, stream>>>(partials, (float*)d_out);
}
